// TransformerLNN_85572928405602
// MI455X (gfx1250) — matrix operand order repaired and checked by reading, not compiled, not run on hardware
//
#include <hip/hip_runtime.h>
#include <hip/hip_bf16.h>
#include <math.h>

// ---------------------------------------------------------------------------
// TransformerLNN for MI455X (gfx1250, wave32, WMMA bf16 16x16x32, f32 accum)
//
// All activations live in GLOBAL memory as bf16 (converted once at each
// producer epilogue); f32 is used only for accumulators, LN statistics and the
// final output. LDS staging is a pure copy and uses the gfx1250 async
// global->LDS path (ASYNCcnt) when the builtin is available.
//
//   x_bf = bf16(x)
//   h    = x_bf @ in_proj^T + b                  (WMMA GEMM)
//   qkv  = h @ qkv_w^T + b                       (WMMA GEMM)
//   ctx  = flash-attention(qkv)                  (WMMA, online softmax)
//   t1   = ctx @ attn_out^T + b + h              (WMMA GEMM, fused residual)
//   h_att= LN1(t1)
//   U    = h_att @ W1x^T + ltc_b1                (xt-half of ltc_w1, hoisted)
//   outs = LTC scan (persistent 32-wave WG, state in LDS, WMMA)
//   y    = LN2(outs + h_att) @ out_w^T + out_b   (WMMA GEMM, f32 out)
//
// Workspace: 6 MB bf16 weights + ~50 MB bf16 activations (~57 MB total).
// ---------------------------------------------------------------------------

typedef __attribute__((ext_vector_type(16))) __bf16       v16bf;
typedef __attribute__((ext_vector_type(8)))  float        v8f;
typedef __attribute__((ext_vector_type(4)))  unsigned int v4u;
typedef __attribute__((ext_vector_type(4)))  int          v4i;

// pointer-to-AS1(v4i) and pointer-to-AS3(v4i) for the async builtin signature
typedef __attribute__((address_space(1))) v4i* gptr_v4i;
typedef __attribute__((address_space(3))) v4i* lptr_v4i;

union BF16x16 { v4u q[2]; v16bf v; };

#if defined(__has_builtin)
# if __has_builtin(__builtin_amdgcn_global_load_async_to_lds_b128)
#  define HAVE_ASYNC_LDS 1
# endif
#endif
#ifndef HAVE_ASYNC_LDS
# define HAVE_ASYNC_LDS 0
#endif

__device__ __forceinline__ unsigned short f2bf(float f) {
  union { float f; unsigned u; } x; x.f = f;
  unsigned u = x.u;
  u += 0x7FFFu + ((u >> 16) & 1u);           // round-to-nearest-even
  return (unsigned short)(u >> 16);
}
__device__ __forceinline__ float bf2f(unsigned short s) {
  union { unsigned u; float f; } x; x.u = ((unsigned)s) << 16; return x.f;
}

// 16B global -> LDS copy; async (ASYNCcnt, no VGPR round trip) when available.
__device__ __forceinline__ void cp16_g2l(const unsigned short* g, unsigned short* l) {
#if HAVE_ASYNC_LDS
  __builtin_amdgcn_global_load_async_to_lds_b128(
      (gptr_v4i)(void*)g, (lptr_v4i)(void*)l, 0, 0);
#else
  *(v4u*)l = *(const v4u*)g;
#endif
}
__device__ __forceinline__ void cp_wait() {
#if HAVE_ASYNC_LDS
# if __has_builtin(__builtin_amdgcn_s_wait_asynccnt)
  __builtin_amdgcn_s_wait_asynccnt(0);
# else
  asm volatile("s_wait_asynccnt 0x0" ::: "memory");
# endif
#endif
}

__device__ __forceinline__ v8f wmma_bf16(v16bf a, v16bf b, v8f c) {
  return __builtin_amdgcn_wmma_f32_16x16x32_bf16(false, a, false, b,
                                                 (short)0, c, false, false);
}

// A-fragment (16x32 bf16) from LDS tile (row-major, rowStride in ushorts).
// lane<16: K=0..15; lane>=16: K=16..31 (contiguous, matches B-frags)
__device__ __forceinline__ v16bf fragA(const unsigned short* base, int rowStride,
                                       int mrow0, int k0) {
  int lane = threadIdx.x & 31;
  int m = mrow0 + (lane & 15), hh = lane >> 4;
  const unsigned short* p = base + m * rowStride + k0 + 16 * hh;
  BF16x16 r;
  r.q[0] = *(const v4u*)p;          // K = k0 + 16h .. +7
  r.q[1] = *(const v4u*)(p + 8);    // K = k0 + 16h + 8 .. +7
  return r.v;
}

// B-fragment (32x16 bf16): W row-major [N][K] (torch weight layout); lane holds
// column n = lane%16, K = 16*(lane/16)..+15 contiguous in a W row.
__device__ __forceinline__ v16bf fragB_g(const unsigned short* W, int K,
                                         int n0, int k0) {
  int lane = threadIdx.x & 31;
  const unsigned short* p = W + (size_t)(n0 + (lane & 15)) * K + k0 + (lane >> 4) * 16;
  BF16x16 r;
  r.q[0] = *(const v4u*)p;
  r.q[1] = *(const v4u*)(p + 8);
  return r.v;
}

__device__ __forceinline__ v16bf fragB_lds(const unsigned short* base, int rowStride,
                                           int n0, int k0) {
  int lane = threadIdx.x & 31;
  const unsigned short* p = base + (n0 + (lane & 15)) * rowStride + k0 + (lane >> 4) * 16;
  BF16x16 r;
  r.q[0] = *(const v4u*)p;
  r.q[1] = *(const v4u*)(p + 8);
  return r.v;
}

// ---------------------------------------------------------------------------
// fp32 -> bf16 conversion (weights; also x). Handles ltc_w1 column split.
// ---------------------------------------------------------------------------
__global__ void convert_bf16_k(const float* __restrict__ src, unsigned short* __restrict__ dst,
                               int rows, int cols, int srcStride, int srcOff) {
  int i = blockIdx.x * 256 + threadIdx.x;
  if (i >= rows * cols) return;
  int r = i / cols, c = i - r * cols;
  dst[i] = f2bf(src[(size_t)r * srcStride + srcOff + c]);
}

// ---------------------------------------------------------------------------
// GEMM: C[M,N] = A_bf16[M,K] @ Wb_bf16[N,K]^T + bias[N] (+ res_bf16[M,N])
// Block 256 thr = 8 waves (4x2), tile 128x64, BK=32. A staged via async->LDS.
// outF32: write f32 (final projection) else bf16.
// ---------------------------------------------------------------------------
#define GBM 128
#define GBN 64
#define GBK 32
#define GAS 40   // A-tile LDS row stride (ushorts): 80B rows, 16B aligned

__global__ __launch_bounds__(256)
void gemm_bf16_k(const unsigned short* __restrict__ A, const unsigned short* __restrict__ Wb,
                 const float* __restrict__ bias, const unsigned short* __restrict__ res,
                 void* __restrict__ Cout, int M, int N, int K, int outF32) {
  __shared__ unsigned short As[GBM * GAS];
  int tid = threadIdx.x;
  int wave = tid >> 5;
  int wm = wave & 3, wn = wave >> 2;
  int m0 = blockIdx.x * GBM;
  int n0 = blockIdx.y * GBN;

  v8f acc[2][2] = {};
  for (int kk = 0; kk < K; kk += GBK) {
    // stage A tile: 128 rows x 32 bf16 = 128 rows x 4 x 16B segments
    for (int i = tid; i < GBM * 4; i += 256) {
      int r = i >> 2, sg = i & 3;
      const unsigned short* gp = A + (size_t)(m0 + r) * K + kk + sg * 8;
      cp16_g2l(gp, As + r * GAS + sg * 8);
      if (kk + GBK < K) __builtin_prefetch(gp + GBK, 0, 0);   // global_prefetch_b8
    }
    cp_wait();          // each wave drains its own ASYNCcnt, then barrier
    __syncthreads();
    v16bf bfr0 = fragB_g(Wb, K, n0 + wn * 32,      kk);
    v16bf bfr1 = fragB_g(Wb, K, n0 + wn * 32 + 16, kk);
    for (int mt = 0; mt < 2; ++mt) {
      v16bf a = fragA(As, GAS, wm * 32 + mt * 16, 0);
      acc[mt][0] = wmma_bf16(a, bfr0, acc[mt][0]);
      acc[mt][1] = wmma_bf16(a, bfr1, acc[mt][1]);
    }
    __syncthreads();
  }
  int lane = tid & 31, hh = lane >> 4, nl = lane & 15;
  for (int mt = 0; mt < 2; ++mt)
    for (int nt = 0; nt < 2; ++nt)
      for (int r = 0; r < 8; ++r) {
        int row = m0 + wm * 32 + mt * 16 + r + 8 * hh;
        int col = n0 + wn * 32 + nt * 16 + nl;
        size_t idx = (size_t)row * N + col;
        float v = acc[mt][nt][r] + bias[col];
        if (res) v += bf2f(res[idx]);
        if (outF32) ((float*)Cout)[idx] = v;
        else        ((unsigned short*)Cout)[idx] = f2bf(v);
      }
}

// ---------------------------------------------------------------------------
// Flash attention. Block = 128 thr = 4 waves; each wave owns 16 query rows of a
// 64-query tile. Keys in 32-wide chunks, online softmax.
// qkv bf16 [B,S,3H]: q @ +0, k @ +H, v @ +2H; per head offset head*64.
// ---------------------------------------------------------------------------
__global__ __launch_bounds__(128)
void attention_k(const unsigned short* __restrict__ qkv, unsigned short* __restrict__ ctx) {
  const int S = 1024, H3 = 1536, H = 512, HD = 64;
  __shared__ unsigned short Ql[64 * 72];      // [q][d]
  __shared__ unsigned short Kl[32 * 72];      // [key][d]
  __shared__ unsigned short Vt[64 * 40];      // [d][key]  (transposed for B-frags)
  __shared__ unsigned short Pl[4][16 * 40];   // per-wave P scratch [q][key]

  int tid = threadIdx.x, wave = tid >> 5, lane = tid & 31;
  int hh = lane >> 4, nl = lane & 15;
  int q0 = blockIdx.x * 64, head = blockIdx.y, bz = blockIdx.z;
  const size_t rowB = (size_t)bz * S;

  // stage Q: 64 rows x 64 bf16 = 64 x 8 x 16B async copies
  for (int i = tid; i < 64 * 8; i += 128) {
    int r = i >> 3, sg = i & 7;
    cp16_g2l(qkv + (rowB + q0 + r) * H3 + head * HD + sg * 8, Ql + r * 72 + sg * 8);
  }
  cp_wait();
  __syncthreads();

  v16bf qf0 = fragA(Ql, 72, wave * 16, 0);
  v16bf qf1 = fragA(Ql, 72, wave * 16, 32);

  v8f o[4] = {};
  float mi[8], li[8];
  for (int r = 0; r < 8; ++r) { mi[r] = -1e30f; li[r] = 0.f; }

  for (int c = 0; c < 32; ++c) {       // 32 chunks x 32 keys
    int k0 = c * 32;
    // K rows: async 16B copies; V: scalar transpose into Vt
    for (int i = tid; i < 32 * 8; i += 128) {
      int r = i >> 3, sg = i & 7;
      cp16_g2l(qkv + (rowB + k0 + r) * H3 + H + head * HD + sg * 8, Kl + r * 72 + sg * 8);
    }
    for (int i = tid; i < 32 * 64; i += 128) {
      int r = i >> 6, d = i & 63;
      Vt[d * 40 + r] = qkv[(rowB + k0 + r) * H3 + 2 * H + head * HD + d];
    }
    cp_wait();
    __syncthreads();

    v8f s0 = {}, s1 = {};
    s0 = wmma_bf16(qf0, fragB_lds(Kl, 72, 0,  0),  s0);
    s0 = wmma_bf16(qf1, fragB_lds(Kl, 72, 0,  32), s0);
    s1 = wmma_bf16(qf0, fragB_lds(Kl, 72, 16, 0),  s1);
    s1 = wmma_bf16(qf1, fragB_lds(Kl, 72, 16, 32), s1);

    float p0[8], p1[8], alpha[8];
    for (int r = 0; r < 8; ++r) {
      float a = s0[r] * 0.125f;    // 1/sqrt(HD)
      float b = s1[r] * 0.125f;
      float mx = fmaxf(a, b);
      mx = fmaxf(mx, __shfl_xor(mx, 1));
      mx = fmaxf(mx, __shfl_xor(mx, 2));
      mx = fmaxf(mx, __shfl_xor(mx, 4));
      mx = fmaxf(mx, __shfl_xor(mx, 8));   // stays within 16-lane half (wave32)
      float mn = fmaxf(mi[r], mx);
      alpha[r] = __expf(mi[r] - mn);
      p0[r] = __expf(a - mn);
      p1[r] = __expf(b - mn);
      float rs = p0[r] + p1[r];
      rs += __shfl_xor(rs, 1); rs += __shfl_xor(rs, 2);
      rs += __shfl_xor(rs, 4); rs += __shfl_xor(rs, 8);
      li[r] = li[r] * alpha[r] + rs;
      mi[r] = mn;
    }
    for (int nt = 0; nt < 4; ++nt)
      for (int r = 0; r < 8; ++r) o[nt][r] *= alpha[r];

    unsigned short* P = Pl[wave];
    for (int r = 0; r < 8; ++r) {
      P[(r + 8 * hh) * 40 + nl]      = f2bf(p0[r]);
      P[(r + 8 * hh) * 40 + 16 + nl] = f2bf(p1[r]);
    }
    v16bf pf = fragA(P, 40, 0, 0);   // same-wave LDS RAW: compiler inserts s_wait_dscnt
    for (int nt = 0; nt < 4; ++nt)
      o[nt] = wmma_bf16(pf, fragB_lds(Vt, 40, nt * 16, 0), o[nt]);
    __syncthreads();
  }

  for (int nt = 0; nt < 4; ++nt)
    for (int r = 0; r < 8; ++r) {
      int q = q0 + wave * 16 + r + 8 * hh;
      ctx[(rowB + q) * H + head * HD + nt * 16 + nl] = f2bf(o[nt][r] / li[r]);
    }
}

// ---------------------------------------------------------------------------
// LayerNorm over H=512 (bf16 in/out, f32 stats): one token per 128-thr block.
// out = LN(x (+x2)) * g + b
// ---------------------------------------------------------------------------
__global__ __launch_bounds__(128)
void layernorm_k(const unsigned short* __restrict__ x, const unsigned short* __restrict__ x2,
                 const float* __restrict__ g, const float* __restrict__ beta,
                 unsigned short* __restrict__ out) {
  const int H = 512;
  size_t tok = blockIdx.x;
  int tid = threadIdx.x;
  float v[4], s = 0.f, ss = 0.f;
  for (int i = 0; i < 4; ++i) {
    float t = bf2f(x[tok * H + tid + i * 128]);
    if (x2) t += bf2f(x2[tok * H + tid + i * 128]);
    v[i] = t; s += t; ss += t * t;
  }
  for (int m = 1; m < 32; m <<= 1) { s += __shfl_xor(s, m); ss += __shfl_xor(ss, m); }
  __shared__ float ws1[4], ws2[4];
  int wave = tid >> 5, lane = tid & 31;
  if (lane == 0) { ws1[wave] = s; ws2[wave] = ss; }
  __syncthreads();
  s  = ws1[0] + ws1[1] + ws1[2] + ws1[3];
  ss = ws2[0] + ws2[1] + ws2[2] + ws2[3];
  float mean = s * (1.f / H);
  float var  = ss * (1.f / H) - mean * mean;
  float inv  = rsqrtf(var + 1e-5f);
  for (int i = 0; i < 4; ++i) {
    int col = tid + i * 128;
    out[tok * H + col] = f2bf((v[i] - mean) * inv * g[col] + beta[col]);
  }
}

// ---------------------------------------------------------------------------
// LTC scan: 1 workgroup, 1024 thr = 32 waves. State LDS-resident (bf16),
// M padded 8->16 (pad rows independent, never read back). Weights L2-resident.
// Per step:  G1 feats_pre = state@W1h^T (+U, precomputed with b1) -> tanh
//            G2 F2 = feats@w2^T + b2
//            G3/4/5 time/g/hn heads; gate combine; write outs + state.
// Each wave owns one 16-col N-tile for every GEMM -> C-frags stay in regs.
// ---------------------------------------------------------------------------
__global__ __launch_bounds__(1024)
void ltc_scan_k(const unsigned short* __restrict__ U,
                const unsigned short* __restrict__ w1h,
                const unsigned short* __restrict__ w2,
                const unsigned short* __restrict__ wt,
                const unsigned short* __restrict__ wg,
                const unsigned short* __restrict__ wh,
                const float* __restrict__ b2v, const float* __restrict__ tbv,
                const float* __restrict__ gbv, const float* __restrict__ hbv,
                unsigned short* __restrict__ outs) {
  const int H = 512, S = 1024, ST = 520;
  __shared__ unsigned short stateb[16 * ST];
  __shared__ unsigned short featsb[16 * ST];
  __shared__ unsigned short f2b[16 * ST];
  int tid = threadIdx.x, wave = tid >> 5, lane = tid & 31;
  int nl = lane & 15, hf = lane >> 4;
  int n = wave * 16 + nl;

  for (int i = tid; i < 16 * ST; i += 1024) stateb[i] = 0;
  __syncthreads();

  float b2n = b2v[n], tbn = tbv[n], gbn = gbv[n], hbn = hbv[n];

  for (int t = 0; t < S; ++t) {
    v8f a1 = {};
    for (int ks = 0; ks < 16; ++ks)
      a1 = wmma_bf16(fragA(stateb, ST, 0, ks * 32),
                     fragB_g(w1h, H, wave * 16, ks * 32), a1);
    for (int r = 0; r < 8; ++r) {
      int m = r + 8 * hf;
      float uv = (m < 8) ? bf2f(U[((size_t)m * S + t) * H + n]) : 0.f;
      featsb[m * ST + n] = f2bf(tanhf(a1[r] + uv));
    }
    __syncthreads();

    v8f a2 = {};
    for (int ks = 0; ks < 16; ++ks)
      a2 = wmma_bf16(fragA(featsb, ST, 0, ks * 32),
                     fragB_g(w2, H, wave * 16, ks * 32), a2);
    for (int r = 0; r < 8; ++r) {
      int m = r + 8 * hf;
      f2b[m * ST + n] = f2bf(a2[r] + b2n);
    }
    __syncthreads();

    v8f at = {}, ag = {}, ah = {};
    for (int ks = 0; ks < 16; ++ks) {
      v16bf af = fragA(f2b, ST, 0, ks * 32);
      at = wmma_bf16(af, fragB_g(wt, H, wave * 16, ks * 32), at);
      ag = wmma_bf16(af, fragB_g(wg, H, wave * 16, ks * 32), ag);
      ah = wmma_bf16(af, fragB_g(wh, H, wave * 16, ks * 32), ah);
    }
    float tt = (float)t;
    for (int r = 0; r < 8; ++r) {
      int m = r + 8 * hf;
      float ft   = 1.f / (1.f + __expf(-(at[r] + tbn)));
      float gate = 1.f / (1.f + __expf(ft * tt));   // sigmoid(-ft*t)
      float hnew = gate * (ag[r] + gbn) + (1.f - gate) * (ah[r] + hbn);
      if (m < 8) outs[((size_t)m * S + t) * H + n] = f2bf(hnew);
      stateb[m * ST + n] = f2bf(hnew);
    }
    __syncthreads();
  }
}

// ---------------------------------------------------------------------------
// Host orchestration
// ---------------------------------------------------------------------------
extern "C" void kernel_launch(void* const* d_in, const int* in_sizes, int n_in,
                              void* d_out, int out_size, void* d_ws, size_t ws_size,
                              hipStream_t stream) {
  (void)in_sizes; (void)n_in; (void)out_size; (void)ws_size;
  const float* x     = (const float*)d_in[0];
  const float* in_w  = (const float*)d_in[1];
  const float* in_b  = (const float*)d_in[2];
  const float* qkv_w = (const float*)d_in[3];
  const float* qkv_b = (const float*)d_in[4];
  const float* ao_w  = (const float*)d_in[5];
  const float* ao_b  = (const float*)d_in[6];
  const float* w1    = (const float*)d_in[7];
  const float* b1    = (const float*)d_in[8];
  const float* w2f   = (const float*)d_in[9];
  const float* b2    = (const float*)d_in[10];
  const float* tw    = (const float*)d_in[11];
  const float* tbv   = (const float*)d_in[12];
  const float* gw    = (const float*)d_in[13];
  const float* gbv   = (const float*)d_in[14];
  const float* hw    = (const float*)d_in[15];
  const float* hbv   = (const float*)d_in[16];
  const float* n1g   = (const float*)d_in[17];
  const float* n1b   = (const float*)d_in[18];
  const float* n2g   = (const float*)d_in[19];
  const float* n2b   = (const float*)d_in[20];
  const float* ow    = (const float*)d_in[21];
  const float* ob    = (const float*)d_in[22];
  float* out = (float*)d_out;

  char* ws = (char*)d_ws;
  const size_t W = 512 * 512;
  unsigned short* wb_in  = (unsigned short*)ws;
  unsigned short* wb_qkv = wb_in  + W;        // 1536x512
  unsigned short* wb_ao  = wb_qkv + 3 * W;
  unsigned short* wb_w1x = wb_ao  + W;
  unsigned short* wb_w1h = wb_w1x + W;
  unsigned short* wb_w2  = wb_w1h + W;
  unsigned short* wb_t   = wb_w2  + W;
  unsigned short* wb_g   = wb_t   + W;
  unsigned short* wb_h   = wb_g   + W;
  unsigned short* wb_o   = wb_h   + W;
  // bf16 activation buffers
  unsigned short* buf0 = wb_o + W;                         // 8192 x 512
  unsigned short* buf1 = buf0 + (size_t)8192 * 512;        // 8192 x 1536
  unsigned short* buf2 = buf1 + (size_t)8192 * 1536;       // 8192 x 512
  unsigned short* buf3 = buf2 + (size_t)8192 * 512;        // 8192 x 512

  auto conv = [&](const float* src, unsigned short* dst, int rows, int cols,
                  int stride, int off) {
    int nelem = rows * cols;
    convert_bf16_k<<<(nelem + 255) / 256, 256, 0, stream>>>(src, dst, rows, cols, stride, off);
  };
  conv(in_w,  wb_in,  512,  512, 512,  0);
  conv(qkv_w, wb_qkv, 1536, 512, 512,  0);
  conv(ao_w,  wb_ao,  512,  512, 512,  0);
  conv(w1,    wb_w1x, 512,  512, 1024, 0);     // columns 0..511   (xt part)
  conv(w1,    wb_w1h, 512,  512, 1024, 512);   // columns 512..1023 (state part)
  conv(w2f,   wb_w2,  512,  512, 512,  0);
  conv(tw,    wb_t,   512,  512, 512,  0);
  conv(gw,    wb_g,   512,  512, 512,  0);
  conv(hw,    wb_h,   512,  512, 512,  0);
  conv(ow,    wb_o,   512,  512, 512,  0);
  conv(x,     buf3,   8192, 512, 512,  0);     // x -> bf16 (buf3, dead after GEMM1)

  // h = x_bf @ in_proj^T + b                               -> buf0
  gemm_bf16_k<<<dim3(64, 8),  256, 0, stream>>>(buf3, wb_in,  in_b,  nullptr, buf0, 8192, 512,  512, 0);
  // qkv = h @ qkv^T + b                                    -> buf1
  gemm_bf16_k<<<dim3(64, 24), 256, 0, stream>>>(buf0, wb_qkv, qkv_b, nullptr, buf1, 8192, 1536, 512, 0);
  // attention                                              -> buf2
  attention_k<<<dim3(16, 8, 8), 128, 0, stream>>>(buf1, buf2);
  // t1 = ctx @ attn_out^T + b + h (fused residual)         -> buf3
  gemm_bf16_k<<<dim3(64, 8),  256, 0, stream>>>(buf2, wb_ao,  ao_b,  buf0,    buf3, 8192, 512,  512, 0);
  // h_att = LN1(t1)                                        -> buf0
  layernorm_k<<<8192, 128, 0, stream>>>(buf3, nullptr, n1g, n1b, buf0);
  // U = h_att @ W1x^T + b1                                 -> buf2
  gemm_bf16_k<<<dim3(64, 8),  256, 0, stream>>>(buf0, wb_w1x, b1,    nullptr, buf2, 8192, 512,  512, 0);
  // LTC scan                                               -> buf3
  ltc_scan_k<<<1, 1024, 0, stream>>>(buf2, wb_w1h, wb_w2, wb_t, wb_g, wb_h,
                                     b2, tbv, gbv, hbv, buf3);
  // LN2(outs + h_att)                                      -> buf1
  layernorm_k<<<8192, 128, 0, stream>>>(buf3, buf0, n2g, n2b, buf1);
  // y = ln2 @ out_w^T + out_b (f32 output)
  gemm_bf16_k<<<dim3(64, 8),  256, 0, stream>>>(buf1, wb_o,   ob,    nullptr, out,  8192, 512,  512, 1);
}
